// KANOnlyTextModel_8014408974698
// MI455X (gfx1250) — compile-verified, matrix operand order checked by reading, not run on hardware
//
#include <hip/hip_runtime.h>

// ---------------------------------------------------------------------------
// KAN (2-layer) as fused f16-WMMA GEMMs for gfx1250 (MI455X, wave32).
//
// Layer l is  y = A(x) @ W_l + affine,  where A(x)[b, d*7+t] is, for input
// dim d:  t<6 -> cubic B-spline basis_t(x[b,d]) on the uniform extended grid,
//         t=6 -> silu(x[b,d]);
// and W_l[d*7+t, o] = t<6 ? coef[d,o,t]*ss[d,o] : sb[d,o].
// Layer1: M=1024 N=128 K=57344; Layer2: M=1024 N=128 K=896.
//
// Weights are converted once to f16 and pre-swizzled into WMMA B-fragment
// order (32x16 K-major tiles, 32 bytes contiguous per lane) so the GEMM loop
// loads B operands with plain coalesced 32B global loads. Basis/silu features
// are computed in registers via the branchless truncated-power form of the
// cardinal cubic B-spline (straight-line VALU only, keeps EXEC all-ones),
// scattered into LDS in A-fragment order, then read back as contiguous
// ds_load_b128 pairs shared by all waves in the block.
// ---------------------------------------------------------------------------

typedef __attribute__((ext_vector_type(16))) _Float16 v16h;
typedef __attribute__((ext_vector_type(8)))  float    v8f;

#define NDIM_OUT 128      // H == V == 128 for both layers
#define WAVES    4        // waves per block (N-slice of 64 columns)
#define THREADS  (WAVES * 32)

// Fragment offset inside a 512-half (32x16) tile, per CDNA5 ISA 7.12.2
// 16-bit layouts. Identical map for A (mn = M row) and B (mn = N col):
//   lane = mn + 16*bit3(k) ; half = (k&7) + 8*bit4(k)
__device__ __forceinline__ int frag_off(int mn, int k) {
    int lane = mn + (((k >> 3) & 1) << 4);
    int h    = (k & 7) + ((k >> 4) << 3);
    return lane * 16 + h;
}

// 7 KAN features for scalar x, fully branchless.
// Grid: K=3, NUM=3, hs=2/3, g[i]=(i-3)*hs-1; basis_j supported on
// [g[j], g[j+4]).  With u=(x-g0)/hs, basis_j(x) = N3(u-j) where N3 is the
// cardinal cubic B-spline, evaluated in truncated-power form:
//   N3(v) = (1/6) * [ v+^3 - 4(v-1)+^3 + 6(v-2)+^3 - 4(v-3)+^3 + (v-4)+^3 ]
// Outside [0,9) all terms cancel to ~0 (u clamped to [-1,10] so the f32
// cancellation noise stays below the f16 quantization of the A fragment).
__device__ __forceinline__ void kan_feats(float x, float f[7]) {
    const float sixth = 1.0f / 6.0f;
    float u = (x + 3.0f) * 1.5f;
    u = fminf(fmaxf(u, -1.0f), 10.0f);
    float q[10];
#pragma unroll
    for (int k = 0; k < 10; ++k) {
        float z = fmaxf(u - (float)k, 0.0f);
        q[k] = z * z * z;
    }
#pragma unroll
    for (int j = 0; j < 6; ++j)
        f[j] = sixth * (q[j] - 4.0f * q[j + 1] + 6.0f * q[j + 2]
                        - 4.0f * q[j + 3] + q[j + 4]);
    // silu via v_exp_f32 + v_rcp_f32 (no IEEE v_div_scale sequence)
    f[6] = x * __builtin_amdgcn_rcpf(1.0f + __expf(-x));
}

// ---------------------------------------------------------------------------
// x[b, s*128+j] = emb[idx[b,s], j]   (float4-vectorized gather)
// ---------------------------------------------------------------------------
__global__ void gather_x(const int* __restrict__ idx,
                         const float* __restrict__ emb,
                         float* __restrict__ x) {
    int tid = blockIdx.x * 256 + threadIdx.x;   // one float4 per thread
    int e = tid << 2;                           // element index in [B, 8192]
    int b = e >> 13;
    int r = e & 8191;
    int s = r >> 7;
    int j = r & 127;
    int row = idx[b * 64 + s];
    *(float4*)(x + e) = *(const float4*)(emb + (row << 7) + j);
}

// ---------------------------------------------------------------------------
// Pack W_l into f16 B-fragment tiles: tile index = (kc*8 + nb), 512 halves
// per tile, element order matches frag_off so GEMM lanes load 32B linearly.
// ---------------------------------------------------------------------------
__global__ void prep_weights(const float* __restrict__ coef,
                             const float* __restrict__ sb,
                             const float* __restrict__ ss,
                             _Float16* __restrict__ wf,
                             int total) {
    int tid = blockIdx.x * 256 + threadIdx.x;
    if (tid >= total) return;
    int tile   = tid >> 9;
    int within = tid & 511;
    int lane = within >> 4;
    int hh   = within & 15;
    int kc = tile >> 3;                 // 32-wide K chunk
    int nb = tile & 7;                  // 16-wide N block
    // invert frag_off: k = (hh&7) | bit4(lane)<<3 | (hh>>3)<<4 ; n = lane&15
    int k = (hh & 7) | (((lane >> 4) & 1) << 3) | ((hh >> 3) << 4);
    int n = lane & 15;
    int r = kc * 32 + k;                // r = d*7 + t
    int d = r / 7;
    int t = r - d * 7;
    int o = nb * 16 + n;
    float val = (t < 6) ? coef[(d * NDIM_OUT + o) * 6 + t] * ss[d * NDIM_OUT + o]
                        : sb[d * NDIM_OUT + o];
    wf[tid] = (_Float16)val;
}

// ---------------------------------------------------------------------------
// Fused KAN layer: features-on-the-fly A, pre-swizzled f16 B, f32 WMMA acc.
// Block = 4 waves, 16-row M tile (blockIdx.x), 64-col N slice (blockIdx.y).
// K chunk = 32 input dims = 224 K elements = 7 WMMAs per wave.
// ---------------------------------------------------------------------------
__global__ __launch_bounds__(THREADS)
void kan_layer_wmma(const float* __restrict__ xsrc, int in_dim,
                    const _Float16* __restrict__ wf,
                    const float* __restrict__ subs, const float* __restrict__ subb,
                    const float* __restrict__ nodes, const float* __restrict__ nodeb,
                    float* __restrict__ out) {
    __shared__ __align__(32) _Float16 sA[7 * 512];   // 7 KB: one 16x224 A chunk

    const int tid  = threadIdx.x;
    const int lane = tid & 31;
    const int wave = tid >> 5;                       // 0..3
    const int nb   = blockIdx.y * WAVES + wave;      // 16-col block 0..7
    const int m0   = blockIdx.x << 4;                // 16 rows
    const int nChunks = in_dim >> 5;                 // 32 dims per chunk

    v8f acc = {0.f, 0.f, 0.f, 0.f, 0.f, 0.f, 0.f, 0.f};

    for (int kc = 0; kc < nChunks; ++kc) {
        const int d0 = kc << 5;
        __syncthreads();                             // done reading previous A
        // ---- stage A chunk: 16 rows x 32 dims -> 16x224 f16 in frag order
#pragma unroll
        for (int i = 0; i < 512 / THREADS; ++i) {
            int p  = tid + i * THREADS;              // 0..511
            int m  = p >> 5;
            int dl = p & 31;
            float x = xsrc[(size_t)(m0 + m) * in_dim + d0 + dl];
            float f[7];
            kan_feats(x, f);
#pragma unroll
            for (int t = 0; t < 7; ++t) {
                int r = dl * 7 + t;                  // K index inside chunk
                sA[((r >> 5) << 9) + frag_off(m, r & 31)] = (_Float16)f[t];
            }
        }
        __syncthreads();                             // A visible to all waves

        const _Float16* wbase = wf + ((size_t)(kc * 7) * 8 + nb) * 512;
        if (kc + 1 < nChunks)                        // speculative L2 prefetch
            __builtin_prefetch(wbase + (size_t)7 * 8 * 512, 0, 1);
#pragma unroll
        for (int c = 0; c < 7; ++c) {
            v16h a = *(const v16h*)(sA + (c << 9) + lane * 16);
            v16h b = *(const v16h*)(wbase + (size_t)c * 8 * 512 + lane * 16);
            acc = __builtin_amdgcn_wmma_f32_16x16x32_f16(
                false, a, false, b, (short)0, acc, false, false);
        }
    }

    // ---- epilogue: y = nodes*(subs*acc + subb) + nodeb, C/D frag layout:
    // VGPR v: lanes 0-15 -> M=v, lanes 16-31 -> M=v+8; N = (lane&15)
    const int n  = nb * 16 + (lane & 15);
    const float s1 = subs[n], b1 = subb[n], s2 = nodes[n], b2 = nodeb[n];
    const int mhi = (lane >> 4) << 3;
#pragma unroll
    for (int v = 0; v < 8; ++v) {
        int m = v + mhi;
        out[(size_t)(m0 + m) * NDIM_OUT + n] = s2 * (s1 * acc[v] + b1) + b2;
    }
}

// ---------------------------------------------------------------------------
extern "C" void kernel_launch(void* const* d_in, const int* in_sizes, int n_in,
                              void* d_out, int out_size, void* d_ws, size_t ws_size,
                              hipStream_t stream) {
    (void)in_sizes; (void)n_in; (void)out_size; (void)ws_size;

    const int*   idx    = (const int*)d_in[0];
    const float* emb    = (const float*)d_in[1];
    const float* coef1  = (const float*)d_in[2];
    const float* sb1    = (const float*)d_in[3];
    const float* ss1    = (const float*)d_in[4];
    const float* subs1  = (const float*)d_in[5];
    const float* subb1  = (const float*)d_in[6];
    const float* nodes1 = (const float*)d_in[7];
    const float* nodeb1 = (const float*)d_in[8];
    const float* coef2  = (const float*)d_in[9];
    const float* sb2    = (const float*)d_in[10];
    const float* ss2    = (const float*)d_in[11];
    const float* subs2  = (const float*)d_in[12];
    const float* subb2  = (const float*)d_in[13];
    const float* nodes2 = (const float*)d_in[14];
    const float* nodeb2 = (const float*)d_in[15];
    float* outp = (float*)d_out;

    // workspace layout (≈49 MB)
    char* ws = (char*)d_ws;
    float*     xbuf = (float*)ws;                         // 1024*8192*4 = 32 MB
    float*     hbuf = (float*)(ws + 33554432);            // 1024*128*4 = 512 KB
    _Float16*  w1   = (_Float16*)(ws + 34078720);         // 57344*128*2 = 14.7 MB
    _Float16*  w2   = (_Float16*)(ws + 48758784);         // 896*128*2 = 229 KB

    const int B = 1024, IN1 = 8192, IN2 = 128;

    // 1) gather embeddings -> x [B, 8192]
    gather_x<<<(B * IN1 / 4) / 256, 256, 0, stream>>>(idx, emb, xbuf);

    // 2) pack both layers' weights into f16 B-fragment tiles
    int tot1 = IN1 * 7 * NDIM_OUT;                        // 7,340,032
    int tot2 = IN2 * 7 * NDIM_OUT;                        // 114,688
    prep_weights<<<(tot1 + 255) / 256, 256, 0, stream>>>(coef1, sb1, ss1, w1, tot1);
    prep_weights<<<(tot2 + 255) / 256, 256, 0, stream>>>(coef2, sb2, ss2, w2, tot2);

    // 3) layer 1: x -> h   (grid: 64 M-tiles x 2 N-slices, 4 waves each)
    dim3 grid(B / 16, NDIM_OUT / (WAVES * 16));
    kan_layer_wmma<<<grid, THREADS, 0, stream>>>(xbuf, IN1, w1,
                                                 subs1, subb1, nodes1, nodeb1, hbuf);
    // 4) layer 2: h -> logits
    kan_layer_wmma<<<grid, THREADS, 0, stream>>>(hbuf, IN2, w2,
                                                 subs2, subb2, nodes2, nodeb2, outp);
}
